// CNNLSTM_39247411151439
// MI455X (gfx1250) — matrix operand order repaired and checked by reading, not compiled, not run on hardware
//
#include <hip/hip_runtime.h>
#include <cstdint>
#include <cstddef>

typedef __attribute__((ext_vector_type(16))) _Float16 v16h;
typedef __attribute__((ext_vector_type(8)))  float    v8f;
typedef unsigned u32x4 __attribute__((ext_vector_type(4)));
typedef int      i32x8 __attribute__((ext_vector_type(8)));
typedef int      i32x4 __attribute__((ext_vector_type(4)));

#if __has_builtin(__builtin_amdgcn_exp2f)
#define DEV_EXP2(x) __builtin_amdgcn_exp2f(x)
#else
#define DEV_EXP2(x) exp2f(x)
#endif
#if __has_builtin(__builtin_amdgcn_rcpf)
#define DEV_RCP(x) __builtin_amdgcn_rcpf(x)
#else
#define DEV_RCP(x) (1.0f / (x))
#endif

#if defined(__HIP_DEVICE_COMPILE__) && __has_builtin(__builtin_amdgcn_tensor_load_to_lds) && \
    __has_builtin(__builtin_amdgcn_s_wait_tensorcnt)
#define USE_TDM 1
#else
#define USE_TDM 0
#endif

__device__ __forceinline__ float dsigm(float x) {
  return DEV_RCP(1.0f + DEV_EXP2(-1.4426950408889634f * x));
}
__device__ __forceinline__ float dtanhf(float x) {
  return 2.0f * dsigm(2.0f * x) - 1.0f;
}

union V16U { v16h v; uint4 q[2]; };
union V8HU { uint4 q; _Float16 h[8]; };

// A-matrix 16x32 f16 tile (ISA layout): lane<16: row=lane, K=kb..kb+7 and kb+16..kb+23;
// lane>=16: same row, kb offset +8. base points at (row0, k0); ld in halves.
__device__ __forceinline__ v16h load_a16(const _Float16* base, int ld, int lane) {
  unsigned row = lane & 15;
  unsigned kb  = (lane >> 4) << 3;             // 0 or 8
  const _Float16* p = base + row * (unsigned)ld + kb;
  V16U u;
  u.q[0] = *(const uint4*)(p);
  u.q[1] = *(const uint4*)(p + 16);
  return u.v;
}

// B-matrix 32x16 f16 tile (ISA layout): lane<16: n=lane, K=k0..k0+7 and k0+16..k0+23;
// lane>=16: n=lane-16, k0 offset +8. W row-major (n, k), ld in halves.
__device__ __forceinline__ v16h load_b16(const _Float16* w, int ld, int n0, int k0, int lane) {
  unsigned n  = n0 + (lane & 15);
  unsigned kk = k0 + ((lane >> 4) << 3);       // 0 or 8
  const _Float16* p = w + n * (unsigned)ld + kk;
  V16U u;
  u.q[0] = *(const uint4*)(p);
  u.q[1] = *(const uint4*)(p + 16);
  return u.v;
}

#if USE_TDM
// TDM: async-copy a 2D tile (16 batch elems x GS gate columns, f16) from gx into LDS.
// D# per CDNA5 ISA ch.8: group0 = {flags/count, lds_addr, global_addr, type=2},
// group1 = {data_size=2B, tensor dims = tile dims (tile never OOB), d0_stride=256}.
// amdgpu-toolchain (clang-23) 6-arg builtin: (u32x4, i32x8, i32x4, i32x4, i32x8, cpol).
__device__ __forceinline__ void tdm_load_gx(const _Float16* gptr, unsigned ldsByte, int GS_) {
  unsigned long long ga = (unsigned long long)(uintptr_t)gptr;
  u32x4 g0 = {1u,                                 // count=1 valid descriptor
              ldsByte,                            // lds_addr (bytes)
              (unsigned)ga,                       // global_addr[31:0]
              (unsigned)(ga >> 32) | (2u << 30)}; // global_addr[56:32] | type=2
  i32x8 g1 = {0x00010000,                         // data_size=1 (2 bytes), no flags
              0x00100000,                         // tensor_dim0[15:0]=16 -> bits[31:16]
              (int)((unsigned)GS_ << 16),         // tensor_dim1[15:0]=GS -> bits[31:16]
              16 << 16,                           // tile_dim0=16 -> bits[31:16]
              GS_,                                // tile_dim1=GS (tile_dim2=0)
              256,                                // tensor_dim0_stride[31:0]
              0, 0};
  i32x4 gz4 = {0, 0, 0, 0};
  i32x8 gz8 = {0, 0, 0, 0, 0, 0, 0, 0};
  __builtin_amdgcn_tensor_load_to_lds(g0, g1, gz4, gz4, gz8, 0);
}
#endif

// ---------------- conv1d(1->128, k=3, pad=1) + ReLU, f32 -> f16 ----------------
__global__ __launch_bounds__(256) void conv_relu(const float* __restrict__ x,
                                                 const float* __restrict__ w,
                                                 const float* __restrict__ bias,
                                                 _Float16* __restrict__ y) {
  unsigned idx = blockIdx.x * 256u + threadIdx.x;  // covers 256*512*128 exactly
  unsigned c  = idx & 127u;
  unsigned bt = idx >> 7;                          // b*512 + t
  unsigned t  = bt & 511u;
  float xm = (t > 0u)   ? x[bt - 1u] : 0.f;
  float x0 = x[bt];
  float xp = (t < 511u) ? x[bt + 1u] : 0.f;
  float v = w[c * 3u] * xm + w[c * 3u + 1u] * x0 + w[c * 3u + 2u] * xp + bias[c];
  y[bt * 128u + c] = (_Float16)(v > 0.f ? v : 0.f);
}

// ------------- pack weights to padded f16 (zero fill), fuse biases -------------
__global__ __launch_bounds__(256) void pack_layer(
    const float* __restrict__ wih, const float* __restrict__ whh,
    const float* __restrict__ bih, const float* __restrict__ bhh,
    _Float16* __restrict__ wihP, _Float16* __restrict__ whhP, float* __restrict__ biasP,
    int fourH, int ins, int KinP, int H, int HPAD) {
  unsigned r = blockIdx.x;
  unsigned tid = threadIdx.x;
  for (unsigned k = tid; k < (unsigned)KinP; k += 256u)
    wihP[r * (unsigned)KinP + k] =
        (r < (unsigned)fourH && k < (unsigned)ins) ? (_Float16)wih[r * (unsigned)ins + k]
                                                   : (_Float16)0.f;
  for (unsigned k = tid; k < (unsigned)HPAD; k += 256u)
    whhP[r * (unsigned)HPAD + k] =
        (r < (unsigned)fourH && k < (unsigned)H) ? (_Float16)whh[r * (unsigned)H + k]
                                                 : (_Float16)0.f;
  if (tid == 0u) biasP[r] = (r < (unsigned)fourH) ? (bih[r] + bhh[r]) : 0.f;
}

// ---- gx = y @ w_ih^T + (b_ih+b_hh); A:(131072 x Kpad) f16, out (T, G, B=256) f16 ----
__global__ __launch_bounds__(32) void gemm_gates(const _Float16* __restrict__ A,
                                                 const _Float16* __restrict__ W,
                                                 const float* __restrict__ bias,
                                                 _Float16* __restrict__ gx,
                                                 int Kpad, int G) {
  int lane = threadIdx.x;              // block = 1 wave
  unsigned m0  = blockIdx.x * 16u;     // row = b*512 + t; a tile never crosses b
  unsigned nt0 = blockIdx.y * 4u;
  v8f z = {0.f, 0.f, 0.f, 0.f, 0.f, 0.f, 0.f, 0.f};
  v8f acc[4] = {z, z, z, z};
  const _Float16* Arow = A + m0 * (unsigned)Kpad;
  for (int k0 = 0; k0 < Kpad; k0 += 32) {
    v16h a = load_a16(Arow + k0, Kpad, lane);
#pragma unroll
    for (int j = 0; j < 4; ++j) {
      v16h b = load_b16(W, Kpad, (int)(nt0 + j) * 16, k0, lane);
      acc[j] = __builtin_amdgcn_wmma_f32_16x16x32_f16(false, a, false, b,
                                                      (short)0, acc[j], false, false);
    }
  }
  unsigned nl = lane & 15;
  unsigned mb = (unsigned)(lane >> 4) * 8u;
  unsigned bb = m0 >> 9;               // batch index (T = 512)
  unsigned tb = (m0 & 511u) + mb;      // time base for r=0
#pragma unroll
  for (int j = 0; j < 4; ++j) {
    unsigned n = (nt0 + j) * 16u + nl;
    float bn = bias[n];
#pragma unroll
    for (int r = 0; r < 8; ++r) {
      unsigned tt = tb + (unsigned)r;
      gx[(tt * (unsigned)G + n) * 256u + bb] = (_Float16)(acc[j][r] + bn);
    }
  }
}

// --------- persistent per-batch-tile LSTM scan: h,c live in LDS, 512 steps ---------
// G = padded global gx stride (NTP*16); GS = NT*16 = LDS gate stride.
// TDM double-buffers next timestep's gx slab into LDS while this step computes.
template <int H, int HPAD, int NT, int G>
__global__ __launch_bounds__(256) void lstm_scan(const _Float16* __restrict__ gx,
                                                 const _Float16* __restrict__ whh,
                                                 _Float16* __restrict__ ynext) {
  constexpr int GS = NT * 16;
  constexpr int KS = HPAD / 32;
  extern __shared__ char smem[];
  float*    sh_g = (float*)smem;                                    // 16*GS f32
  float*    sh_c = (float*)(smem + (size_t)16 * GS * 4);            // 16*H f32
  _Float16* sh_h = (_Float16*)(smem + (size_t)16 * GS * 4 + (size_t)16 * H * 4);
  _Float16* sh_x0 = (_Float16*)(smem + (size_t)16 * GS * 4 + (size_t)16 * H * 4 +
                                (size_t)16 * HPAD * 2);             // 16*GS f16 (staging A)
  _Float16* sh_x1 = sh_x0 + (size_t)16 * GS;                        // 16*GS f16 (staging B)

  const int tid  = threadIdx.x;        // 256 threads = 8 waves
  const int lane = tid & 31;
  const int wave = tid >> 5;
  const unsigned b0 = blockIdx.x * 16u;

  for (int i = tid; i < 16 * HPAD; i += 256) sh_h[i] = (_Float16)0.f;
  for (int i = tid; i < 16 * H;    i += 256) sh_c[i] = 0.f;
#if USE_TDM
  if (wave == 0) {
    tdm_load_gx(gx + b0, (unsigned)(uintptr_t)(void*)sh_x0, GS);
    __builtin_amdgcn_s_wait_tensorcnt(0);
  }
#endif
  __syncthreads();

  const unsigned nl = lane & 15;
  const unsigned mb = (unsigned)(lane >> 4) * 8u;

  for (int t = 0; t < 512; ++t) {
    // Preload all A-tiles of h into registers; reused across this wave's N-tiles.
    v16h a[KS];
#pragma unroll
    for (int kk = 0; kk < KS; ++kk) a[kk] = load_a16(sh_h + kk * 32, HPAD, lane);

#if USE_TDM
    const _Float16* xb = (t & 1) ? sh_x1 : sh_x0;   // [n][16 batch] f16, staged by TDM
#else
    const _Float16* grow = gx + (unsigned)t * (unsigned)G * 256u + b0 + mb;
#endif
    for (int nt = wave; nt < NT; nt += 8) {
      unsigned n = (unsigned)nt * 16u + nl;
      V8HU ci;
#if USE_TDM
      ci.q = *(const uint4*)(xb + n * 16u + mb);    // C init = staged gx, 16B LDS load
#else
      ci.q = *(const uint4*)(grow + n * 256u);      // C init = gx, 16B global load
#endif
      v8f c;
#pragma unroll
      for (int r = 0; r < 8; ++r) c[r] = (float)ci.h[r];
#pragma unroll
      for (int kk = 0; kk < KS; ++kk) {
        v16h b = load_b16(whh, HPAD, nt * 16, kk * 32, lane);
        c = __builtin_amdgcn_wmma_f32_16x16x32_f16(false, a[kk], false, b,
                                                   (short)0, c, false, false);
      }
#pragma unroll
      for (int r = 0; r < 8; ++r) sh_g[(mb + (unsigned)r) * GS + n] = c[r];
    }
    __syncthreads();

#if USE_TDM
    // Kick off next timestep's gx slab into the other buffer; overlaps the gate math.
    if (wave == 0 && t < 511) {
      _Float16* nxt = (t & 1) ? sh_x0 : sh_x1;
      tdm_load_gx(gx + (unsigned)(t + 1) * (unsigned)G * 256u + b0,
                  (unsigned)(uintptr_t)(void*)nxt, GS);
    }
#endif

    // Elementwise LSTM cell update; also zero-fills the K-padding of ynext.
    for (int idx = tid; idx < 16 * HPAD; idx += 256) {
      int m = idx / HPAD;
      int j = idx - m * HPAD;
      _Float16 hv = (_Float16)0.f;
      if (j < H) {
        float gi = sh_g[m * GS + j];
        float gf = sh_g[m * GS + H + j];
        float gc = sh_g[m * GS + 2 * H + j];
        float go = sh_g[m * GS + 3 * H + j];
        float cp = sh_c[m * H + j];
        float cn = dsigm(gf) * cp + dsigm(gi) * dtanhf(gc);
        sh_c[m * H + j] = cn;
        hv = (_Float16)(dsigm(go) * dtanhf(cn));
      }
      sh_h[idx] = hv;
      ynext[((b0 + (unsigned)m) * 512u + (unsigned)t) * (unsigned)HPAD + (unsigned)j] = hv;
    }
#if USE_TDM
    if (wave == 0) __builtin_amdgcn_s_wait_tensorcnt(0);
#endif
    __syncthreads();
  }
}

// ---------------- final FC: out[b] = y5[b, T-1, :70] . fc_w + fc_b ----------------
__global__ __launch_bounds__(256) void fc_out(const _Float16* __restrict__ y5,
                                              const float* __restrict__ fw,
                                              const float* __restrict__ fb,
                                              float* __restrict__ out) {
  unsigned b = threadIdx.x;
  if (b < 256u) {
    const _Float16* p = y5 + (b * 512u + 511u) * 96u;  // HPAD of layer 5 = 96
    float s = fb[0];
    for (int j = 0; j < 70; ++j) s += (float)p[j] * fw[j];
    out[b] = s;
  }
}

extern "C" void kernel_launch(void* const* d_in, const int* in_sizes, int n_in,
                              void* d_out, int out_size, void* d_ws, size_t ws_size,
                              hipStream_t stream) {
  (void)in_sizes; (void)n_in; (void)out_size; (void)ws_size;
  const float* x      = (const float*)d_in[0];
  const float* conv_w = (const float*)d_in[1];
  const float* conv_b = (const float*)d_in[2];
  const float *wih[5], *whh[5], *bih[5], *bhh[5];
  for (int l = 0; l < 5; ++l) {
    wih[l] = (const float*)d_in[3 + 4 * l];
    whh[l] = (const float*)d_in[4 + 4 * l];
    bih[l] = (const float*)d_in[5 + 4 * l];
    bhh[l] = (const float*)d_in[6 + 4 * l];
  }
  const float* fc_w = (const float*)d_in[23];
  const float* fc_b = (const float*)d_in[24];
  float* out = (float*)d_out;

  static const int INS[5]  = {128, 200, 160, 130, 100};
  static const int KINP[5] = {128, 224, 160, 160, 128};  // = HPAD of previous layer
  static const int HH[5]   = {200, 160, 130, 100, 70};
  static const int HP[5]   = {224, 160, 160, 128, 96};
  static const int NTP[5]  = {52, 40, 36, 28, 20};       // ceil(4H/16) padded to mult of 4
  static const int GL[5]   = {832, 640, 576, 448, 320};  // NTP*16 (gx stride)

  // Workspace carve (~337 MB): yA,yB f16 ping-pong; gx (T,G,B) f16; packed weights.
  char* ws = (char*)d_ws;
  size_t off = 0;
  auto carve = [&](size_t bytes) -> void* {
    void* p = ws + off;
    off += (bytes + 255) & ~(size_t)255;
    return p;
  };
  _Float16* yA    = (_Float16*)carve((size_t)131072 * 224 * 2);
  _Float16* yB    = (_Float16*)carve((size_t)131072 * 224 * 2);
  _Float16* gxb   = (_Float16*)carve((size_t)512 * 832 * 256 * 2);
  _Float16* wihP  = (_Float16*)carve((size_t)832 * 224 * 2);
  _Float16* whhP  = (_Float16*)carve((size_t)832 * 224 * 2);
  float*    biasP = (float*)   carve((size_t)832 * 4);

  conv_relu<<<65536, 256, 0, stream>>>(x, conv_w, conv_b, yA);

  for (int l = 0; l < 5; ++l) {
    int fourH = 4 * HH[l];
    int NProws = NTP[l] * 16;
    pack_layer<<<NProws, 256, 0, stream>>>(wih[l], whh[l], bih[l], bhh[l],
                                           wihP, whhP, biasP,
                                           fourH, INS[l], KINP[l], HH[l], HP[l]);
    const _Float16* src = (l % 2 == 0) ? yA : yB;
    _Float16* dst       = (l % 2 == 0) ? yB : yA;
    dim3 gg(8192, (unsigned)(NTP[l] / 4));
    gemm_gates<<<gg, 32, 0, stream>>>(src, wihP, biasP, gxb, KINP[l], GL[l]);
    int GS = ((4 * HH[l] + 15) / 16) * 16;   // actual gate-tile columns (LDS stride)
    size_t shb = (size_t)16 * GS * 4 + (size_t)16 * HH[l] * 4 + (size_t)16 * HP[l] * 2 +
                 (size_t)2 * 16 * GS * 2;    // + two TDM staging buffers
    switch (l) {
      case 0: lstm_scan<200, 224, 50, 832><<<16, 256, shb, stream>>>(gxb, whhP, dst); break;
      case 1: lstm_scan<160, 160, 40, 640><<<16, 256, shb, stream>>>(gxb, whhP, dst); break;
      case 2: lstm_scan<130, 160, 33, 576><<<16, 256, shb, stream>>>(gxb, whhP, dst); break;
      case 3: lstm_scan<100, 128, 25, 448><<<16, 256, shb, stream>>>(gxb, whhP, dst); break;
      case 4: lstm_scan< 70,  96, 18, 320><<<16, 256, shb, stream>>>(gxb, whhP, dst); break;
    }
  }
  fc_out<<<1, 256, 0, stream>>>(yB, fc_w, fc_b, out);
}
